// APPNPNet_46213848105787
// MI455X (gfx1250) — compile-verified, matrix operand order checked by reading, not run on hardware
//
#include <hip/hip_runtime.h>
#include <hip/hip_bf16.h>

typedef __attribute__((ext_vector_type(16))) __bf16 v16bf;
typedef __attribute__((ext_vector_type(8)))  float  v8f;

#define F_IN  512
#define F_H   128
#define F_OUT 64
#define ALPHA 0.1f

// ---------------------------------------------------------------------------
// Small utility kernels
// ---------------------------------------------------------------------------
__global__ void set_ones_kernel(float* __restrict__ p, int n) {
    int i = blockIdx.x * blockDim.x + threadIdx.x;
    if (i < n) p[i] = 1.0f;   // self-loop contributes 1 to every node's degree
}

__global__ void deg_scatter_kernel(const int* __restrict__ col, float* __restrict__ deg, int E) {
    int e = blockIdx.x * blockDim.x + threadIdx.x;
    if (e < E) unsafeAtomicAdd(&deg[col[e]], 1.0f);
}

__global__ void dinv_kernel(float* __restrict__ dinv, int n) {
    int i = blockIdx.x * blockDim.x + threadIdx.x;
    if (i < n) {
        float d = dinv[i];
        dinv[i] = (d > 0.0f) ? rsqrtf(d) : 0.0f;
    }
}

// ---------------------------------------------------------------------------
// GEMM1: H[N,128] = relu(X[N,512] @ W1[512,128] + b1)   (bf16 WMMA, f32 accum)
// Block = 256 threads = 8 waves; each wave owns a 16-row strip x all 128 cols.
// ---------------------------------------------------------------------------
__global__ void gemm1_kernel(const float* __restrict__ X, const float* __restrict__ W1,
                             const float* __restrict__ b1, float* __restrict__ H, int N) {
    extern __shared__ char smem_raw[];
    v16bf* sB = (v16bf*)smem_raw;                 // 8 ntiles * 32 lanes fragments (8 KB)

    const int tid   = threadIdx.x;
    const int lane  = tid & 31;
    const int wave  = tid >> 5;
    const int nlow  = lane & 15;
    const int khalf = lane >> 4;
    const int blockRow = blockIdx.x * 128;
    const bool full = (blockRow + 128) <= N;      // 781 of 782 blocks: no guards

    v8f acc[8];
    const v8f vz = {0.f, 0.f, 0.f, 0.f, 0.f, 0.f, 0.f, 0.f};
#pragma unroll
    for (int i = 0; i < 8; ++i) acc[i] = vz;

    int arow = blockRow + wave * 16 + nlow;
    if (!full && arow >= N) arow = N - 1;         // clamp; store guarded later
    const float* xbase = X + (size_t)arow * F_IN + khalf * 8;

    for (int ks = 0; ks < 16; ++ks) {
        const int k0 = ks * 32;
        // ---- cooperative fill of B fragments into LDS (fragment-major) ----
        {
            const int nt = tid >> 5;              // 0..7 n-tile
            const int kb = khalf * 8;
            const float* wp = W1 + (size_t)k0 * F_H + nt * 16 + nlow;
            v16bf frag;
#pragma unroll
            for (int i = 0; i < 16; ++i) {
                const int kk = (i < 8) ? (kb + i) : (kb + 8 + i);   // runs [kb,kb+8) and [kb+16,kb+24)
                frag[i] = (__bf16)wp[(size_t)kk * F_H];
            }
            sB[tid] = frag;
        }
        __syncthreads();

        // ---- A fragment: two contiguous 8-float runs, cvt to bf16 ----
        const float* xr = xbase + k0;
        v16bf afrag;
#pragma unroll
        for (int i = 0; i < 8; ++i) afrag[i] = (__bf16)xr[i];
#pragma unroll
        for (int i = 0; i < 8; ++i) afrag[8 + i] = (__bf16)xr[16 + i];

#pragma unroll
        for (int nt = 0; nt < 8; ++nt) {
            v16bf bfrag = sB[nt * 32 + lane];
            acc[nt] = __builtin_amdgcn_wmma_f32_16x16x32_bf16(
                false, afrag, false, bfrag, (short)0, acc[nt], false, false);
        }
        __syncthreads();
    }

    // ---- epilogue: +bias, relu, store ----
    const int mhi = khalf * 8;
    float* hout = H + (size_t)(blockRow + wave * 16 + mhi) * F_H + nlow;
    if (full) {
#pragma unroll
        for (int nt = 0; nt < 8; ++nt) {
            const float bias = b1[nt * 16 + nlow];
#pragma unroll
            for (int r = 0; r < 8; ++r) {
                float v = acc[nt][r] + bias;
                hout[(size_t)r * F_H + nt * 16] = v > 0.f ? v : 0.f;
            }
        }
    } else {
#pragma unroll
        for (int nt = 0; nt < 8; ++nt) {
            const float bias = b1[nt * 16 + nlow];
#pragma unroll
            for (int r = 0; r < 8; ++r) {
                const int row = blockRow + wave * 16 + mhi + r;
                if (row < N) {
                    float v = acc[nt][r] + bias;
                    H[(size_t)row * F_H + nt * 16 + nlow] = v > 0.f ? v : 0.f;
                }
            }
        }
    }
}

// ---------------------------------------------------------------------------
// GEMM2: out[N,64] = H[N,128] @ W2[128,64] + b2  -> written to x0 AND z(d_out)
// ---------------------------------------------------------------------------
__global__ void gemm2_kernel(const float* __restrict__ H, const float* __restrict__ W2,
                             const float* __restrict__ b2, float* __restrict__ x0,
                             float* __restrict__ z, int N) {
    extern __shared__ char smem_raw[];
    v16bf* sB = (v16bf*)smem_raw;                 // 4 ntiles * 32 lanes (4 KB)

    const int tid   = threadIdx.x;
    const int lane  = tid & 31;
    const int wave  = tid >> 5;
    const int nlow  = lane & 15;
    const int khalf = lane >> 4;
    const int blockRow = blockIdx.x * 128;
    const bool full = (blockRow + 128) <= N;

    v8f acc[4];
    const v8f vz = {0.f, 0.f, 0.f, 0.f, 0.f, 0.f, 0.f, 0.f};
#pragma unroll
    for (int i = 0; i < 4; ++i) acc[i] = vz;

    int arow = blockRow + wave * 16 + nlow;
    if (!full && arow >= N) arow = N - 1;
    const float* hbase = H + (size_t)arow * F_H + khalf * 8;

    for (int ks = 0; ks < 4; ++ks) {
        const int k0 = ks * 32;
        if (tid < 128) {                          // 4 ntiles * 32 lanes fragments
            const int nt = tid >> 5;
            const int kb = khalf * 8;
            const float* wp = W2 + (size_t)k0 * F_OUT + nt * 16 + nlow;
            v16bf frag;
#pragma unroll
            for (int i = 0; i < 16; ++i) {
                const int kk = (i < 8) ? (kb + i) : (kb + 8 + i);
                frag[i] = (__bf16)wp[(size_t)kk * F_OUT];
            }
            sB[tid] = frag;
        }
        __syncthreads();

        const float* hr = hbase + k0;
        v16bf afrag;
#pragma unroll
        for (int i = 0; i < 8; ++i) afrag[i] = (__bf16)hr[i];
#pragma unroll
        for (int i = 0; i < 8; ++i) afrag[8 + i] = (__bf16)hr[16 + i];

#pragma unroll
        for (int nt = 0; nt < 4; ++nt) {
            v16bf bfrag = sB[nt * 32 + lane];
            acc[nt] = __builtin_amdgcn_wmma_f32_16x16x32_bf16(
                false, afrag, false, bfrag, (short)0, acc[nt], false, false);
        }
        __syncthreads();
    }

    const int mhi = khalf * 8;
    if (full) {
        const size_t base = (size_t)(blockRow + wave * 16 + mhi) * F_OUT + nlow;
#pragma unroll
        for (int nt = 0; nt < 4; ++nt) {
            const float bias = b2[nt * 16 + nlow];
#pragma unroll
            for (int r = 0; r < 8; ++r) {
                const float v = acc[nt][r] + bias;
                const size_t off = base + (size_t)r * F_OUT + nt * 16;
                x0[off] = v;
                z[off]  = v;
            }
        }
    } else {
#pragma unroll
        for (int nt = 0; nt < 4; ++nt) {
            const float bias = b2[nt * 16 + nlow];
#pragma unroll
            for (int r = 0; r < 8; ++r) {
                const int row = blockRow + wave * 16 + mhi + r;
                if (row < N) {
                    const float v = acc[nt][r] + bias;
                    const size_t off = (size_t)row * F_OUT + nt * 16 + nlow;
                    x0[off] = v;
                    z[off]  = v;
                }
            }
        }
    }
}

// ---------------------------------------------------------------------------
// Propagation step kernels. agg is initialized with the self-loop term
// (dinv^2 * z), then edges scatter, then z = 0.9*agg + 0.1*x0.
// ---------------------------------------------------------------------------
__global__ void agg_init_kernel(float4* __restrict__ agg, const float4* __restrict__ z,
                                const float* __restrict__ dinv, int N) {
    const long i = (long)blockIdx.x * blockDim.x + threadIdx.x;
    if (i >= (long)N * 16) return;
    const float di = dinv[(int)(i >> 4)];
    const float sn = di * di;
    const float4 zz = z[i];
    agg[i] = make_float4(sn * zz.x, sn * zz.y, sn * zz.z, sn * zz.w);
}

__global__ void edge_scatter_kernel(const int* __restrict__ row, const int* __restrict__ col,
                                    const float* __restrict__ dinv, const float* __restrict__ z,
                                    float* __restrict__ agg, int E) {
    const int g    = blockIdx.x * blockDim.x + threadIdx.x;
    const int e    = g >> 5;
    const int lane = g & 31;
    if (e >= E) return;
    const int r = row[e];
    const int c = col[e];
    const float w = dinv[r] * dinv[c];
    const float2 v = ((const float2*)(z + (size_t)r * F_OUT))[lane];
    float* ap = agg + (size_t)c * F_OUT + lane * 2;
    unsafeAtomicAdd(ap,     w * v.x);
    unsafeAtomicAdd(ap + 1, w * v.y);
}

__global__ void combine_kernel(const float4* __restrict__ agg, const float4* __restrict__ x0,
                               float4* __restrict__ z, long n4) {
    const long i = (long)blockIdx.x * blockDim.x + threadIdx.x;
    if (i >= n4) return;
    const float4 a = agg[i];
    const float4 x = x0[i];
    float4 o;
    o.x = (1.f - ALPHA) * a.x + ALPHA * x.x;
    o.y = (1.f - ALPHA) * a.y + ALPHA * x.y;
    o.z = (1.f - ALPHA) * a.z + ALPHA * x.z;
    o.w = (1.f - ALPHA) * a.w + ALPHA * x.w;
    z[i] = o;
}

// ---------------------------------------------------------------------------
// Row-wise log_softmax over 64 channels; one wave32 per row, 2 ch per lane.
// ---------------------------------------------------------------------------
__global__ void logsoftmax_kernel(float* __restrict__ z, int N) {
    const int g    = blockIdx.x * blockDim.x + threadIdx.x;
    const int row  = g >> 5;
    const int lane = g & 31;
    if (row >= N) return;
    float* zr = z + (size_t)row * F_OUT;
    const float a = zr[lane];
    const float b = zr[32 + lane];
    float m = fmaxf(a, b);
#pragma unroll
    for (int off = 16; off > 0; off >>= 1) m = fmaxf(m, __shfl_xor(m, off, 32));
    float s = __expf(a - m) + __expf(b - m);
#pragma unroll
    for (int off = 16; off > 0; off >>= 1) s += __shfl_xor(s, off, 32);
    const float lse = m + __logf(s);
    zr[lane]      = a - lse;
    zr[32 + lane] = b - lse;
}

// ---------------------------------------------------------------------------
extern "C" void kernel_launch(void* const* d_in, const int* in_sizes, int n_in,
                              void* d_out, int out_size, void* d_ws, size_t ws_size,
                              hipStream_t stream) {
    const float* X  = (const float*)d_in[0];
    const int*   EI = (const int*)d_in[1];
    const float* W1 = (const float*)d_in[2];
    const float* b1 = (const float*)d_in[3];
    const float* W2 = (const float*)d_in[4];
    const float* b2 = (const float*)d_in[5];
    float* out = (float*)d_out;                    // [N,64]; also serves as z

    const int N = in_sizes[0] / F_IN;
    const int E = in_sizes[1] / 2;
    const int* row = EI;                           // edge_index[0] = source
    const int* col = EI + E;                       // edge_index[1] = target

    // workspace layout
    char* ws = (char*)d_ws;
    float* dinv = (float*)ws;
    size_t off = (((size_t)N * 4) + 255) & ~(size_t)255;
    float* x0  = (float*)(ws + off);
    float* agg = x0  + (size_t)N * F_OUT;
    float* Hbuf = agg + (size_t)N * F_OUT;

    const int T = 256;

    // degree (with self-loop) -> dinv
    set_ones_kernel<<<(N + T - 1) / T, T, 0, stream>>>(dinv, N);
    deg_scatter_kernel<<<(E + T - 1) / T, T, 0, stream>>>(col, dinv, E);
    dinv_kernel<<<(N + T - 1) / T, T, 0, stream>>>(dinv, N);

    // MLP encoder via WMMA
    const int gblocks = (N + 127) / 128;
    gemm1_kernel<<<gblocks, T, 8 * 32 * sizeof(v16bf), stream>>>(X, W1, b1, Hbuf, N);
    gemm2_kernel<<<gblocks, T, 4 * 32 * sizeof(v16bf), stream>>>(Hbuf, W2, b2, x0, out, N);

    // APPNP propagation (K=16); z/agg are L2-resident (25.6 MB each)
    const long n4 = (long)N * 16;
    const int zg = (int)((n4 + T - 1) / T);
    const int sg = (int)(((long)E * 32 + T - 1) / T);
    for (int k = 0; k < 16; ++k) {
        agg_init_kernel<<<zg, T, 0, stream>>>((float4*)agg, (const float4*)out, dinv, N);
        edge_scatter_kernel<<<sg, T, 0, stream>>>(row, col, dinv, out, agg, E);
        combine_kernel<<<zg, T, 0, stream>>>((const float4*)agg, (const float4*)x0,
                                             (float4*)out, n4);
    }

    // final log-softmax
    logsoftmax_kernel<<<(int)(((long)N * 32 + T - 1) / T), T, 0, stream>>>(out, N);
}
